// NearestNeighborAttention_32787780337664
// MI455X (gfx1250) — matrix operand order repaired and checked by reading, not compiled, not run on hardware
//
#include <hip/hip_runtime.h>
#include <hip/hip_bf16.h>
#include <stdint.h>

typedef __attribute__((ext_vector_type(16))) _Float16 v16h;
typedef __attribute__((ext_vector_type(8)))  float    v8f;

#define NPTS  4096
#define DIM   1024
#define HEADS 16
#define HD    64
#define KNNK  16

// ---- CDNA5 async copy: global -> LDS, tracked by ASYNCcnt (CDNA5 ISA) ----
__device__ __forceinline__ void async_copy_b128(uint32_t lds_addr, const void* gptr) {
  asm volatile("global_load_async_to_lds_b128 %0, %1, off"
               :: "v"(lds_addr), "v"((uint64_t)(uintptr_t)gptr)
               : "memory");
}
__device__ __forceinline__ uint32_t lds_addr_of(const void* p) {
  // generic shared pointer = {aperture_hi, lds_offset_lo}; LDS addr = addr[31:0]
  return (uint32_t)(uintptr_t)p;
}

// ---------------- f32 -> f16 conversion ----------------
__global__ void cvt_f16_kernel(const float* __restrict__ src, _Float16* __restrict__ dst, int n) {
  int i = blockIdx.x * blockDim.x + threadIdx.x;
  if (i < n) dst[i] = (_Float16)src[i];
}

// ---------------- mask compaction (ordered nonzero) ----------------
__global__ void mask_count_kernel(const unsigned char* __restrict__ mask, int* __restrict__ counts) {
  if (threadIdx.x != 0) return;
  int b = blockIdx.x;                       // 128 chunks of 1024
  const unsigned char* p = mask + b * 1024;
  int c = 0;
  for (int i = 0; i < 1024; ++i) c += (p[i] != 0);
  counts[b] = c;
}

__global__ void mask_write_kernel(const unsigned char* __restrict__ mask, const int* __restrict__ counts,
                                  int* __restrict__ active) {
  __shared__ int offs[128];
  if (threadIdx.x == 0) {
    int s = 0;
    for (int i = 0; i < 128; ++i) { offs[i] = s; s += counts[i]; }
  }
  __syncthreads();
  int b = threadIdx.x;                      // blockDim.x == 128
  int o = offs[b];
  const unsigned char* p = mask + b * 1024;
  for (int i = 0; i < 1024; ++i) if (p[i]) active[o++] = b * 1024 + i;
}

__global__ void coords_kernel(const int* __restrict__ active, float4* __restrict__ coords) {
  int i = blockIdx.x * blockDim.x + threadIdx.x;
  if (i >= NPTS) return;
  int idx = active[i];
  float c0 = (float)(idx >> 12);            // VOL = (32, 64, 64)
  float c1 = (float)((idx >> 6) & 63);
  float c2 = (float)(idx & 63);
  float sq = c0 * c0 + c1 * c1 + c2 * c2;
  coords[i] = make_float4(c0, c1, c2, sq);
}

// ---------------- KNN (top-17 smallest d2, drop self) ----------------
__global__ __launch_bounds__(256) void knn_kernel(const float4* __restrict__ coords, int* __restrict__ nbr) {
  __shared__ float4 cs[NPTS];               // 64 KB of the 320 KB WGP LDS
  int t = threadIdx.x;
  for (int i = t; i < NPTS; i += 256) cs[i] = coords[i];
  __syncthreads();
  int i = blockIdx.x * 256 + t;
  float4 ci = cs[i];
  float bd[KNNK + 1];
  int   bi[KNNK + 1];
  for (int m = 0; m <= KNNK; ++m) { bd[m] = 3.4e38f; bi[m] = 0; }
  for (int j = 0; j < NPTS; ++j) {
    float4 cj = cs[j];
    float d2 = ci.w + cj.w - 2.0f * (ci.x * cj.x + ci.y * cj.y + ci.z * cj.z);
    if (d2 < bd[KNNK]) {                    // strict '<' => stable ties (lower index wins)
      int m = KNNK;
      while (m > 0 && bd[m - 1] > d2) { bd[m] = bd[m - 1]; bi[m] = bi[m - 1]; --m; }
      bd[m] = d2; bi[m] = j;
    }
  }
  for (int kk = 0; kk < KNNK; ++kk) nbr[i * KNNK + kk] = bi[kk + 1];  // skip self (d2 == 0)
}

// ---------------- fused QKV GEMM: y = x @ W^T via v_wmma_f32_16x16x32_f16 ----------------
// Workgroup tile 128(M) x 64(N); 8 waves in 4x2 grid; wave tile 32x32 (2x2 WMMA).
// Double-buffered LDS stages filled by GLOBAL_LOAD_ASYNC_TO_LDS_B128 (ASYNCcnt-tracked),
// overlapping the global fetch of stage i+1 with the WMMAs of stage i.
__global__ __launch_bounds__(256) void gemm_qkv_kernel(const _Float16* __restrict__ xh,
                                                       const _Float16* __restrict__ wh,
                                                       float* __restrict__ qbuf,
                                                       float* __restrict__ kbuf,
                                                       float* __restrict__ vbuf) {
  __shared__ __align__(16) _Float16 As[2][128][48];  // x rows, K-tile 32, padded to 48 halves
  __shared__ __align__(16) _Float16 Bs[2][64][48];   // W rows (B[k][n] = W[n][k])
  const int mat = blockIdx.z;
  const _Float16* W = wh + (size_t)mat * DIM * DIM;
  float* outp = (mat == 0) ? qbuf : (mat == 1) ? kbuf : vbuf;
  const int Mbase = blockIdx.x * 128;
  const int Nbase = blockIdx.y * 64;
  const int t    = threadIdx.x;
  const int lane = t & 31;
  const int wv   = t >> 5;
  const int wm   = wv >> 1, wn = wv & 1;
  const int g    = lane >> 4, l16 = lane & 15;

  v8f c00 = {}, c01 = {}, c10 = {}, c11 = {};

  // staging assignment: A = 32B/thread (2 x b128), B = 16B/thread (1 x b128)
  const int arow = t >> 1, ahs = (t & 1) * 16;
  const int bcol = t >> 2, bhs = (t & 3) * 8;
  const _Float16* agp = xh + (size_t)(Mbase + arow) * DIM + ahs;
  const _Float16* bgp = W  + (size_t)(Nbase + bcol) * DIM + bhs;
  uint32_t aL0[2], aL1[2], bL0[2];
#pragma unroll
  for (int s = 0; s < 2; ++s) {
    aL0[s] = lds_addr_of(&As[s][arow][ahs]);
    aL1[s] = lds_addr_of(&As[s][arow][ahs + 8]);
    bL0[s] = lds_addr_of(&Bs[s][bcol][bhs]);
  }

  // prologue: stage 0
  async_copy_b128(aL0[0], agp);
  async_copy_b128(aL1[0], agp + 8);
  async_copy_b128(bL0[0], bgp);

  for (int k0 = 0; k0 < DIM; k0 += 32) {
    const int s = (k0 >> 5) & 1;
    if (k0 + 32 < DIM) {                       // issue next stage into the other buffer
      async_copy_b128(aL0[s ^ 1], agp + k0 + 32);
      async_copy_b128(aL1[s ^ 1], agp + k0 + 40);
      async_copy_b128(bL0[s ^ 1], bgp + k0 + 32);
      asm volatile("s_wait_asynccnt 3" ::: "memory");  // older 3 (current stage) done
    } else {
      asm volatile("s_wait_asynccnt 0" ::: "memory");
    }
    __syncthreads();                            // all waves' copies for stage s visible

    // Fragment loads per ISA 7.12.2:
    //  A (16x32 f16): lane m=l16, VGPR0-3 = K[g*8 .. g*8+7], VGPR4-7 = K[16+g*8 ..]
    //  B (32x16 f16): lane n=l16, VGPR0-3 = K[g*8 .. g*8+7], VGPR4-7 = K[16+g*8 ..]
    union FragU { uint4 u[2]; v16h h; };
    FragU fa0, fa1, fb0, fb1;
    const int ra = wm * 32 + l16;
    fa0.u[0] = *(const uint4*)(&As[s][ra][g * 8]);
    fa0.u[1] = *(const uint4*)(&As[s][ra][16 + g * 8]);
    fa1.u[0] = *(const uint4*)(&As[s][ra + 16][g * 8]);
    fa1.u[1] = *(const uint4*)(&As[s][ra + 16][16 + g * 8]);
    const int cb = wn * 32 + l16;
    fb0.u[0] = *(const uint4*)(&Bs[s][cb][g * 8]);
    fb0.u[1] = *(const uint4*)(&Bs[s][cb][16 + g * 8]);
    fb1.u[0] = *(const uint4*)(&Bs[s][cb + 16][g * 8]);
    fb1.u[1] = *(const uint4*)(&Bs[s][cb + 16][16 + g * 8]);

    c00 = __builtin_amdgcn_wmma_f32_16x16x32_f16(false, fa0.h, false, fb0.h, (short)0, c00, false, false);
    c01 = __builtin_amdgcn_wmma_f32_16x16x32_f16(false, fa0.h, false, fb1.h, (short)0, c01, false, false);
    c10 = __builtin_amdgcn_wmma_f32_16x16x32_f16(false, fa1.h, false, fb0.h, (short)0, c10, false, false);
    c11 = __builtin_amdgcn_wmma_f32_16x16x32_f16(false, fa1.h, false, fb1.h, (short)0, c11, false, false);
    __syncthreads();                            // stage s may be overwritten next iter
  }

  // C/D layout: lane n=l16, VGPR r -> M = r + g*8
  for (int r = 0; r < 8; ++r) {
    int m0 = Mbase + wm * 32 + g * 8 + r;
    int n0 = Nbase + wn * 32 + l16;
    outp[(size_t)m0 * DIM + n0]              = c00[r];
    outp[(size_t)m0 * DIM + n0 + 16]         = c01[r];
    outp[(size_t)(m0 + 16) * DIM + n0]       = c10[r];
    outp[(size_t)(m0 + 16) * DIM + n0 + 16]  = c11[r];
  }
}

// ---------------- metric = mean over heads of k ----------------
__global__ void metric_kernel(const float* __restrict__ kbuf, float* __restrict__ metric, int total) {
  int i = blockIdx.x * blockDim.x + threadIdx.x;
  if (i >= total) return;
  int dd = i & (HD - 1);
  size_t pn = (size_t)(i >> 6);
  const float* row = kbuf + pn * DIM + dd;
  float s = 0.f;
#pragma unroll
  for (int h = 0; h < HEADS; ++h) s += row[h * HD];
  metric[i] = s * (1.0f / 16.0f);
}

// ---------------- neighbor attention: one block per (b, n) point ----------------
__global__ __launch_bounds__(256) void attn_kernel(const float* __restrict__ qbuf,
                                                   const float* __restrict__ kbuf,
                                                   const float* __restrict__ vbuf,
                                                   const int* __restrict__ nbr,
                                                   float* __restrict__ out) {
  __shared__ float qs[DIM];
  __shared__ float sc[HEADS][KNNK];
  __shared__ float at[HEADS][KNNK];
  __shared__ int   nb[KNNK];
  int pn = blockIdx.x;                 // 0 .. B*N-1
  int b  = pn >> 12;
  int n  = pn & (NPTS - 1);
  int t  = threadIdx.x;
  const float* qrow = qbuf + (size_t)pn * DIM;
  for (int i = t; i < DIM; i += 256) qs[i] = qrow[i];
  if (t < KNNK) nb[t] = nbr[n * KNNK + t];
  __syncthreads();

  int h = t >> 4, kk = t & 15;         // 16 heads x 16 neighbors = 256 threads
  const float* krow = kbuf + ((size_t)b * NPTS + nb[kk]) * DIM + h * HD;
  float acc = 0.f;
#pragma unroll
  for (int d = 0; d < HD; ++d) acc += qs[h * HD + d] * krow[d];
  sc[h][kk] = acc * 0.125f;            // 1/sqrt(64)
  __syncthreads();

  float mx = -3.4e38f;
#pragma unroll
  for (int j = 0; j < KNNK; ++j) mx = fmaxf(mx, sc[h][j]);
  float den = 0.f;
#pragma unroll
  for (int j = 0; j < KNNK; ++j) den += __expf(sc[h][j] - mx);
  at[h][kk] = __expf(sc[h][kk] - mx) / den;
  __syncthreads();

  int e0 = t * 4;                      // 4 contiguous output elems per thread
  int hh = e0 >> 6;
  float o0 = 0.f, o1 = 0.f, o2 = 0.f, o3 = 0.f;
  for (int j = 0; j < KNNK; ++j) {
    float w = at[hh][j];
    const float4 vr = *(const float4*)(vbuf + ((size_t)b * NPTS + nb[j]) * DIM + e0);
    o0 += w * vr.x; o1 += w * vr.y; o2 += w * vr.z; o3 += w * vr.w;
  }
  *(float4*)(out + (size_t)pn * DIM + e0) = make_float4(o0, o1, o2, o3);
}

extern "C" void kernel_launch(void* const* d_in, const int* in_sizes, int n_in,
                              void* d_out, int out_size, void* d_ws, size_t ws_size,
                              hipStream_t stream) {
  const float*         x    = (const float*)d_in[0];
  const unsigned char* mask = (const unsigned char*)d_in[1];
  const float*         Wq   = (const float*)d_in[2];
  const float*         Wk   = (const float*)d_in[3];
  const float*         Wv   = (const float*)d_in[4];
  const int B = in_sizes[0] / (NPTS * DIM);   // 2
  const int M = B * NPTS;                     // 8192

  char* ws = (char*)d_ws;
  size_t off = 0;
  _Float16* xh = (_Float16*)(ws + off); off += (size_t)M * DIM * sizeof(_Float16);
  _Float16* wh = (_Float16*)(ws + off); off += (size_t)3 * DIM * DIM * sizeof(_Float16);
  float* qbuf  = (float*)(ws + off);    off += (size_t)M * DIM * sizeof(float);
  float* kbuf  = (float*)(ws + off);    off += (size_t)M * DIM * sizeof(float);
  float* vbuf  = (float*)(ws + off);    off += (size_t)M * DIM * sizeof(float);
  int* active  = (int*)(ws + off);      off += (size_t)NPTS * sizeof(int);
  float4* crd  = (float4*)(ws + off);   off += (size_t)NPTS * sizeof(float4);
  int* nbr     = (int*)(ws + off);      off += (size_t)NPTS * KNNK * sizeof(int);
  int* counts  = (int*)(ws + off);      off += 128 * sizeof(int);

  float* outp   = (float*)d_out;
  float* metric = outp + (size_t)M * DIM;

  int nx = M * DIM;
  cvt_f16_kernel<<<(nx + 255) / 256, 256, 0, stream>>>(x, xh, nx);
  int nw = DIM * DIM;
  cvt_f16_kernel<<<(nw + 255) / 256, 256, 0, stream>>>(Wq, wh, nw);
  cvt_f16_kernel<<<(nw + 255) / 256, 256, 0, stream>>>(Wk, wh + (size_t)DIM * DIM, nw);
  cvt_f16_kernel<<<(nw + 255) / 256, 256, 0, stream>>>(Wv, wh + (size_t)2 * DIM * DIM, nw);

  mask_count_kernel<<<128, 64, 0, stream>>>(mask, counts);
  mask_write_kernel<<<1, 128, 0, stream>>>(mask, counts, active);
  coords_kernel<<<(NPTS + 255) / 256, 256, 0, stream>>>(active, crd);
  knn_kernel<<<NPTS / 256, 256, 0, stream>>>(crd, nbr);

  dim3 ggrid(M / 128, DIM / 64, 3);
  gemm_qkv_kernel<<<ggrid, 256, 0, stream>>>(xh, wh, qbuf, kbuf, vbuf);

  int mtotal = M * HD;
  metric_kernel<<<(mtotal + 255) / 256, 256, 0, stream>>>(kbuf, metric, mtotal);
  attn_kernel<<<M, 256, 0, stream>>>(qbuf, kbuf, vbuf, nbr, outp);
}
